// GlobalAttention_88218628260314
// MI455X (gfx1250) — compile-verified, matrix operand order checked by reading, not run on hardware
//
#include <hip/hip_runtime.h>
#include <hip/hip_bf16.h>

// Problem constants (from reference)
#define B_   16
#define T_   1024
#define S_   1024
#define ENC_ 768
#define DEC_ 768
#define KCAT 1536   // DEC+ENC

typedef __bf16 bf16;
typedef __attribute__((ext_vector_type(16))) __bf16 v16bf;
typedef __attribute__((ext_vector_type(8)))  __bf16 v8bf;
typedef __attribute__((ext_vector_type(8)))  float  v8f;

// Scheduling fence: forbid the machine scheduler from sinking prefetch loads
// past the WMMA block (otherwise it serializes loads against v_wmma and the
// register allocator collapses the double buffer).
#if defined(__has_builtin)
#if __has_builtin(__builtin_amdgcn_sched_barrier)
#define SCHED_BARRIER() __builtin_amdgcn_sched_barrier(0)
#endif
#endif
#ifndef SCHED_BARRIER
#define SCHED_BARRIER() asm volatile("" ::: "memory")
#endif

// ---------------------------------------------------------------------------
// bf16 A/B fragment loader for v_wmma_f32_16x16x32_bf16 (wave32).
// 16-bit operand layout (ISA 7.12.2): lane L owns row/col (L&15); lanes 0-15
// hold K slabs {k0+0..7, k0+16..23}, lanes 16-31 hold {k0+8..15, k0+24..31}.
// => two contiguous 16-byte loads per lane per 32-K chunk.
// ---------------------------------------------------------------------------
__device__ __forceinline__ v16bf load_frag(const bf16* __restrict__ row,
                                           int k0, int half) {
  union { v16bf v; v8bf h[2]; } u;
  u.h[0] = *(const v8bf*)(row + k0 + half * 8);
  u.h[1] = *(const v8bf*)(row + k0 + 16 + half * 8);
  return u.v;
}

__device__ __forceinline__ v8f wmma_bf16(v16bf a, v16bf b, v8f c) {
  return __builtin_amdgcn_wmma_f32_16x16x32_bf16(
      /*neg_a=*/false, a, /*neg_b=*/false, b,
      /*c_mod=*/(short)0, c, /*reuse_a=*/false, /*reuse_b=*/false);
}

// ---------------------------------------------------------------------------
// Small helper kernels
// ---------------------------------------------------------------------------
__global__ void k_zero(float* p, int n) {
  int i = blockIdx.x * blockDim.x + threadIdx.x;
  if (i < n) p[i] = 0.0f;
}

__global__ void k_cvt_bf16(const float* __restrict__ src, bf16* __restrict__ dst, int n) {
  for (int i = blockIdx.x * blockDim.x + threadIdx.x; i < n;
       i += gridDim.x * blockDim.x)
    dst[i] = (bf16)src[i];
}

// ctxT[b][d][s] = ctx[b][s][d]   (so GEMM-2's B operand is row-contiguous)
__global__ void k_transpose_ctx(const float* __restrict__ ctx, bf16* __restrict__ ctxT) {
  int i = blockIdx.x * blockDim.x + threadIdx.x;   // over B*ENC*S = 12.58M
  const int total = B_ * ENC_ * S_;
  if (i >= total) return;
  int s = i % S_;
  int r = i / S_;
  int d = r % ENC_;
  int b = r / ENC_;
  ctxT[i] = (bf16)ctx[((long)b * S_ + s) * ENC_ + d];
}

// gate = sigmoid(colsum)
__global__ void k_gate(const float* __restrict__ colsum, float* __restrict__ gate, int n) {
  int i = blockIdx.x * blockDim.x + threadIdx.x;
  if (i < n) gate[i] = 1.0f / (1.0f + __expf(-colsum[i]));
}

// av *= gate * (1/1024) in place (output region), plus bf16 copy [b,t,s] for GEMM-2
__global__ __launch_bounds__(256) void k_scale_av(float* __restrict__ out_av,
                                                  const float* __restrict__ gate,
                                                  bf16* __restrict__ avgb) {
  const int t = blockIdx.x, b = blockIdx.y;
  const long obase = ((long)t * B_ + b) * S_;     // out layout [T,B,S]
  const long abase = ((long)b * T_ + t) * S_;     // ws  layout [B,T,S]
  for (int s = threadIdx.x; s < S_; s += 256) {
    float v = out_av[obase + s] * gate[b * S_ + s] * (1.0f / (float)S_);
    out_av[obase + s] = v;
    avgb[abase + s] = (bf16)v;
  }
}

// ---------------------------------------------------------------------------
// K1: align = inputs @ ctx^T  (K=768), softmax over S in registers,
//     write un-gated av to output region, atomic colsum accumulation.
// Grid: (T/16, B), 256 threads = 8 waves; wave w covers S cols [w*128, +128).
// Double-buffered fragments; sched_barrier pins chunk-(kc+1) loads above the
// chunk-kc WMMA block so the matrix pipe overlaps ~18 in-flight loads.
// ---------------------------------------------------------------------------
__global__ __launch_bounds__(256)
void k_align_softmax(const bf16* __restrict__ inb, const bf16* __restrict__ ctxb,
                     float* __restrict__ out_av, float* __restrict__ colsum) {
  __shared__ float red[128];      // 8 waves x 16 rows
  __shared__ float rowstat[16];

  const int b = blockIdx.y, t0 = blockIdx.x * 16;
  const int tid = threadIdx.x, lane = tid & 31, wave = tid >> 5;
  const int n = lane & 15, half = lane >> 4, mg = half * 8;
  const int sBase = wave * 128;
  const int NC = ENC_ / 32;       // 24 K-chunks

  v8f acc[8] = {};

  const bf16* arow  = inb  + ((long)b * T_ + t0 + n) * ENC_;
  const bf16* brow0 = ctxb + ((long)b * S_ + sBase + n) * ENC_;  // + j*16*ENC_

  v16bf abuf[2];
  v16bf bbuf[2][8];
  abuf[0] = load_frag(arow, 0, half);
#pragma unroll
  for (int j = 0; j < 8; ++j)
    bbuf[0][j] = load_frag(brow0 + (long)j * 16 * ENC_, 0, half);

#pragma unroll
  for (int kc = 0; kc < NC; ++kc) {
    const int cur = kc & 1, nxt = cur ^ 1;
    const int k0 = kc * 32;
    if (kc < NC - 1) {
      abuf[nxt] = load_frag(arow, k0 + 32, half);
#pragma unroll
      for (int j = 0; j < 8; ++j)
        bbuf[nxt][j] = load_frag(brow0 + (long)j * 16 * ENC_, k0 + 32, half);
    }
    SCHED_BARRIER();
#pragma unroll
    for (int j = 0; j < 8; ++j)
      acc[j] = wmma_bf16(abuf[cur], bbuf[cur][j], acc[j]);
    SCHED_BARRIER();
  }

  // ---- row max (acc[j][r] = row mg+r, col sBase+j*16+n) ----
  float p[8];
#pragma unroll
  for (int r = 0; r < 8; ++r) {
    float m = acc[0][r];
#pragma unroll
    for (int j = 1; j < 8; ++j) m = fmaxf(m, acc[j][r]);
    p[r] = m;
  }
#pragma unroll
  for (int off = 1; off < 16; off <<= 1)
#pragma unroll
    for (int r = 0; r < 8; ++r) p[r] = fmaxf(p[r], __shfl_xor(p[r], off, 32));
  if (n == 0)
#pragma unroll
    for (int r = 0; r < 8; ++r) red[wave * 16 + mg + r] = p[r];
  __syncthreads();
  if (tid < 16) {
    float m = red[tid];
    for (int w = 1; w < 8; ++w) m = fmaxf(m, red[w * 16 + tid]);
    rowstat[tid] = m;
  }
  __syncthreads();
  float rmax[8];
#pragma unroll
  for (int r = 0; r < 8; ++r) rmax[r] = rowstat[mg + r];

  // ---- exp + row sum ----
  float ps[8] = {};
#pragma unroll
  for (int j = 0; j < 8; ++j)
#pragma unroll
    for (int r = 0; r < 8; ++r) {
      float e = __expf(acc[j][r] - rmax[r]);
      acc[j][r] = e;
      ps[r] += e;
    }
#pragma unroll
  for (int off = 1; off < 16; off <<= 1)
#pragma unroll
    for (int r = 0; r < 8; ++r) ps[r] += __shfl_xor(ps[r], off, 32);
  if (n == 0)
#pragma unroll
    for (int r = 0; r < 8; ++r) red[wave * 16 + mg + r] = ps[r];
  __syncthreads();
  if (tid < 16) {
    float s = 0.0f;
    for (int w = 0; w < 8; ++w) s += red[w * 16 + tid];
    rowstat[tid] = 1.0f / s;
  }
  __syncthreads();
  float rinv[8];
#pragma unroll
  for (int r = 0; r < 8; ++r) rinv[r] = rowstat[mg + r];

  // ---- normalize, write av (pre-gate) to out[T,B,S], column partial sums ----
  float csum[8];
#pragma unroll
  for (int j = 0; j < 8; ++j) {
    float cs = 0.0f;
#pragma unroll
    for (int r = 0; r < 8; ++r) {
      float av = acc[j][r] * rinv[r];
      out_av[((long)(t0 + mg + r) * B_ + b) * S_ + sBase + j * 16 + n] = av;
      cs += av;
    }
    csum[j] = cs;
  }
#pragma unroll
  for (int j = 0; j < 8; ++j) csum[j] += __shfl_xor(csum[j], 16, 32); // merge halves
  if (half == 0)
#pragma unroll
    for (int j = 0; j < 8; ++j)
      atomicAdd(&colsum[b * S_ + sBase + j * 16 + n], csum[j]);
}

// ---------------------------------------------------------------------------
// K3: c = av_g @ ctx  (K=S=1024, N=ENC=768), scalar affine, bf16 store.
// Grid: (T/16, B); wave w covers 96 cols (6 n-tiles). Double-buffered.
// ---------------------------------------------------------------------------
__global__ __launch_bounds__(256)
void k_ctx_gemm(const bf16* __restrict__ avgb, const bf16* __restrict__ ctxTb,
                const float* __restrict__ wsc, const float* __restrict__ bsc,
                bf16* __restrict__ cb) {
  const int b = blockIdx.y, t0 = blockIdx.x * 16;
  const int tid = threadIdx.x, lane = tid & 31, wave = tid >> 5;
  const int n = lane & 15, half = lane >> 4, mg = half * 8;
  const int nBase = wave * 96;
  const int NC = S_ / 32;         // 32 K-chunks

  v8f acc[6] = {};
  const bf16* arow  = avgb  + ((long)b * T_ + t0 + n) * S_;
  const bf16* brow0 = ctxTb + ((long)b * ENC_ + nBase + n) * S_;  // + j*16*S_

  v16bf abuf[2];
  v16bf bbuf[2][6];
  abuf[0] = load_frag(arow, 0, half);
#pragma unroll
  for (int j = 0; j < 6; ++j)
    bbuf[0][j] = load_frag(brow0 + (long)j * 16 * S_, 0, half);

#pragma unroll
  for (int kc = 0; kc < NC; ++kc) {
    const int cur = kc & 1, nxt = cur ^ 1;
    const int k0 = kc * 32;
    if (kc < NC - 1) {
      abuf[nxt] = load_frag(arow, k0 + 32, half);
#pragma unroll
      for (int j = 0; j < 6; ++j)
        bbuf[nxt][j] = load_frag(brow0 + (long)j * 16 * S_, k0 + 32, half);
    }
    SCHED_BARRIER();
#pragma unroll
    for (int j = 0; j < 6; ++j)
      acc[j] = wmma_bf16(abuf[cur], bbuf[cur][j], acc[j]);
    SCHED_BARRIER();
  }

  const float w = wsc[0], bs = bsc[0];
#pragma unroll
  for (int j = 0; j < 6; ++j) {
    const int d = nBase + j * 16 + n;
#pragma unroll
    for (int r = 0; r < 8; ++r)
      cb[((long)b * T_ + t0 + mg + r) * ENC_ + d] = (bf16)(acc[j][r] * w + bs);
  }
}

// ---------------------------------------------------------------------------
// K4: h = [c || inputs] @ W_out^T + b_out, PReLU, write [T,B,DEC].
// K=1536 as one double-buffered loop: 24 chunks from c, then 24 from inputs.
// W_out rows are the B-operand (already [out,in] row-major => row-contiguous).
// ---------------------------------------------------------------------------
__global__ __launch_bounds__(256)
void k_final(const bf16* __restrict__ cb, const bf16* __restrict__ inb,
             const bf16* __restrict__ wb, const float* __restrict__ bo,
             const float* __restrict__ pa, float* __restrict__ out_h) {
  const int b = blockIdx.y, t0 = blockIdx.x * 16;
  const int tid = threadIdx.x, lane = tid & 31, wave = tid >> 5;
  const int n = lane & 15, half = lane >> 4, mg = half * 8;
  const int nBase = wave * 96;

  v8f acc[6] = {};
  const bf16* arowC = cb  + ((long)b * T_ + t0 + n) * DEC_;
  const bf16* arowI = inb + ((long)b * T_ + t0 + n) * DEC_;
  const bf16* wrow0 = wb + (long)(nBase + n) * KCAT;              // + j*16*KCAT

  v16bf abuf[2];
  v16bf bbuf[2][6];
  abuf[0] = load_frag(arowC, 0, half);
#pragma unroll
  for (int j = 0; j < 6; ++j)
    bbuf[0][j] = load_frag(wrow0 + (long)j * 16 * KCAT, 0, half);

#pragma unroll
  for (int kc = 0; kc < 48; ++kc) {
    const int cur = kc & 1, nxt = cur ^ 1;
    const int wk0 = kc * 32;
    if (kc < 47) {
      const int kn = kc + 1;
      const bf16* ar = (kn < 24) ? arowC : arowI;
      const int ak0 = (kn < 24) ? kn * 32 : (kn - 24) * 32;
      abuf[nxt] = load_frag(ar, ak0, half);
#pragma unroll
      for (int j = 0; j < 6; ++j)
        bbuf[nxt][j] = load_frag(wrow0 + (long)j * 16 * KCAT, wk0 + 32, half);
    }
    SCHED_BARRIER();
#pragma unroll
    for (int j = 0; j < 6; ++j)
      acc[j] = wmma_bf16(abuf[cur], bbuf[cur][j], acc[j]);
    SCHED_BARRIER();
  }

  const float alpha = pa[0];
#pragma unroll
  for (int j = 0; j < 6; ++j) {
    const int d = nBase + j * 16 + n;
    const float bias = bo[d];
#pragma unroll
    for (int r = 0; r < 8; ++r) {
      float h = acc[j][r] + bias;
      h = (h >= 0.0f) ? h : alpha * h;
      out_h[((long)(t0 + mg + r) * B_ + b) * DEC_ + d] = h;
    }
  }
}

// ---------------------------------------------------------------------------
// Host-side launcher
// ---------------------------------------------------------------------------
extern "C" void kernel_launch(void* const* d_in, const int* in_sizes, int n_in,
                              void* d_out, int out_size, void* d_ws, size_t ws_size,
                              hipStream_t stream) {
  (void)in_sizes; (void)n_in; (void)out_size; (void)ws_size;

  const float* inputs  = (const float*)d_in[0];
  const float* context = (const float*)d_in[1];
  const float* W_out   = (const float*)d_in[2];
  const float* b_out   = (const float*)d_in[3];
  const float* w_scale = (const float*)d_in[4];
  const float* b_scale = (const float*)d_in[5];
  const float* prelu_a = (const float*)d_in[6];

  float* out_h  = (float*)d_out;                       // [T,B,DEC]
  float* out_av = out_h + (long)T_ * B_ * DEC_;        // [T,B,S]

  char* w = (char*)d_ws;
  float* colsum  = (float*)w;  w += (long)B_ * S_ * 4;
  float* gate    = (float*)w;  w += (long)B_ * S_ * 4;
  bf16*  in_bf   = (bf16*)w;   w += (long)B_ * T_ * DEC_ * 2;
  bf16*  ctx_bf  = (bf16*)w;   w += (long)B_ * S_ * ENC_ * 2;
  bf16*  ctxT_bf = (bf16*)w;   w += (long)B_ * ENC_ * S_ * 2;
  bf16*  w_bf    = (bf16*)w;   w += (long)DEC_ * KCAT * 2;
  bf16*  avg_bf  = (bf16*)w;   w += (long)B_ * T_ * S_ * 2;
  bf16*  c_bf    = (bf16*)w;   w += (long)B_ * T_ * ENC_ * 2;

  // Phase 0: zero colsum, bf16 conversions, context transpose
  k_zero<<<64, 256, 0, stream>>>(colsum, B_ * S_);
  k_cvt_bf16<<<4096, 256, 0, stream>>>(inputs,  in_bf,  B_ * T_ * DEC_);
  k_cvt_bf16<<<4096, 256, 0, stream>>>(context, ctx_bf, B_ * S_ * ENC_);
  k_cvt_bf16<<<1024, 256, 0, stream>>>(W_out,   w_bf,   DEC_ * KCAT);
  k_transpose_ctx<<<(B_ * ENC_ * S_) / 256, 256, 0, stream>>>(context, ctxT_bf);

  // Phase 1: scores + softmax + column sums
  k_align_softmax<<<dim3(T_ / 16, B_), 256, 0, stream>>>(in_bf, ctx_bf, out_av, colsum);

  // Phase 2: gate, scale av in place (pooled == 1/S since softmax rows sum to 1)
  k_gate<<<64, 256, 0, stream>>>(colsum, gate, B_ * S_);
  k_scale_av<<<dim3(T_, B_), 256, 0, stream>>>(out_av, gate, avg_bf);

  // Phase 3: c = av_g @ ctx, scalar affine
  k_ctx_gemm<<<dim3(T_ / 16, B_), 256, 0, stream>>>(avg_bf, ctxT_bf, w_scale, b_scale, c_bf);

  // Phase 4: final linear + PReLU
  k_final<<<dim3(T_ / 16, B_), 256, 0, stream>>>(c_bf, in_bf, w_bf, b_out, prelu_a, out_h);
}
